// MeshEncoderDecoder_30056181137888
// MI455X (gfx1250) — matrix operand order repaired and checked by reading, not compiled, not run on hardware
//
#include <hip/hip_runtime.h>
#include <hip/hip_bf16.h>
#include <stdint.h>
#include <stddef.h>

#define NNODES 50000
#define NEDGES 800000
#define NETOT  (NEDGES + NNODES)   // reference adds one self-loop per node
#define NHEADS 8

typedef __attribute__((ext_vector_type(16))) __bf16 bf16x16;
typedef __attribute__((ext_vector_type(8)))  float  floatx8;

// ---------------------------------------------------------------- utilities

__device__ __forceinline__ unsigned short f2bf(float f) {
  // round-to-nearest-even float32 -> bf16 (bit-level, no __bf16 scalar conv)
  unsigned int u = __float_as_uint(f);
  u += 0x7FFFu + ((u >> 16) & 1u);
  return (unsigned short)(u >> 16);
}

__device__ __forceinline__ void atomicMaxF(float* addr, float val) {
  // monotonic mapping trick; values are finite (leaky-relu of finite sums)
  if (val >= 0.0f) atomicMax((int*)addr, __float_as_int(val));
  else             atomicMin((unsigned int*)addr, __float_as_uint(val));
}

__device__ __forceinline__ void edge_sd(const int* __restrict__ ei, int e, int& s, int& d) {
  if (e < NEDGES) { s = ei[e]; d = ei[NEDGES + e]; }
  else            { s = e - NEDGES; d = s; }       // self-loops
}

// ---------------------------------------------------------------- fill

__global__ void k_fill(float* __restrict__ p, float v, int n) {
  int i = blockIdx.x * blockDim.x + threadIdx.x;
  if (i < n) p[i] = v;
}

// ------------------------------------------------- activation f32 -> bf16
// Writes [NNODES, Kp] bf16, fusing the decoder skip-concat and zero K-padding.

__global__ void k_to_bf16(const float* __restrict__ X1, int c1,
                          const float* __restrict__ X2, int c2,
                          unsigned short* __restrict__ Y, int Kp) {
  int i = blockIdx.x * blockDim.x + threadIdx.x;
  int total = NNODES * Kp;
  if (i >= total) return;
  int n = i / Kp, k = i - n * Kp;
  float v = 0.0f;
  if (k < c1)           v = X1[(size_t)n * c1 + k];
  else if (k < c1 + c2) v = X2[(size_t)n * c2 + (k - c1)];
  Y[i] = f2bf(v);
}

// ------------------------------------------------- weight pack f32 -> bf16
// W is [K, Nc] row-major. Packed layout: tile (kt, nt) = 32 lanes x 16 bf16,
// lane L: column n = nt*16 + (L&15); element i: k = kt*32 + (L>>4)*8 + (i&7) + (i>>3)*16.
// This matches the CDNA5 16-bit B-matrix (32x16 KxN) VGPR striping.

__global__ void k_pack_w(const float* __restrict__ W,
                         unsigned short* __restrict__ Wp,
                         int K, int Kp, int Nc) {
  int i = blockIdx.x * blockDim.x + threadIdx.x;
  int total = Kp * Nc;
  if (i >= total) return;
  int tile = i >> 9;          // 512 elems per 32x16 tile
  int rem  = i & 511;
  int lane = rem >> 4;
  int ii   = rem & 15;
  int ntn  = Nc >> 4;
  int kt   = tile / ntn;
  int nt   = tile - kt * ntn;
  int k = kt * 32 + (lane >> 4) * 8 + (ii & 7) + (ii >> 3) * 16;
  int n = nt * 16 + (lane & 15);
  float v = (k < K) ? W[(size_t)k * Nc + n] : 0.0f;
  Wp[i] = f2bf(v);
}

// ------------------------------------------------- WMMA GEMM
// C[M,Nc] = A[M,Kp](bf16) x W(packed bf16) + bias, optional ReLU.
// One wave per 16x16 output tile; K-loop of v_wmma_f32_16x16x32_bf16.
// A fragment per ISA: lane-group g holds K = g*8..g*8+7 and 16+g*8..16+g*8+7.

__global__ void k_gemm_bf16_wmma(const unsigned short* __restrict__ A,
                                 const unsigned short* __restrict__ Wp,
                                 const float* __restrict__ bias,
                                 float* __restrict__ C,
                                 int Kp, int Nc, int ntiles, int relu) {
  int wave = blockIdx.x * (blockDim.x >> 5) + (threadIdx.x >> 5);
  if (wave >= ntiles) return;
  int lane = threadIdx.x & 31;
  int ntn  = Nc >> 4;
  int tm   = wave / ntn;
  int tn   = wave - tm * ntn;
  int g    = lane >> 4;
  int l15  = lane & 15;
  int row  = tm * 16 + l15;

  floatx8 acc = {0.f, 0.f, 0.f, 0.f, 0.f, 0.f, 0.f, 0.f};
  int ktiles = Kp >> 5;
  const unsigned short* arow = A + (size_t)row * Kp + g * 8;

  for (int kt = 0; kt < ktiles; ++kt) {
    union { bf16x16 v; uint4 u[2]; } a, b;
    const unsigned short* ap = arow + kt * 32;
    a.u[0] = *(const uint4*)(ap);
    a.u[1] = *(const uint4*)(ap + 16);
    const unsigned short* bp = Wp + (((size_t)(kt * ntn + tn)) << 9) + lane * 16;
    b.u[0] = *(const uint4*)(bp);
    b.u[1] = *(const uint4*)(bp + 8);
    if (kt + 1 < ktiles) __builtin_prefetch(arow + (kt + 1) * 32, 0, 1);
    acc = __builtin_amdgcn_wmma_f32_16x16x32_bf16(
        false, a.v, false, b.v, (short)0, acc, false, false);
  }

  int col = tn * 16 + l15;
  float bv = bias ? bias[col] : 0.0f;
#pragma unroll
  for (int r = 0; r < 8; ++r) {
    int rr = tm * 16 + g * 8 + r;           // C/D layout: VGPR r <-> M = r + g*8
    float v = acc[r] + bv;
    if (relu) v = fmaxf(v, 0.0f);
    C[(size_t)rr * Nc + col] = v;
  }
}

// ------------------------------------------------- LayerNorm (wave per node)
// Y = LN(X + prebias) * g + b, optional ReLU. In-place safe (row-private).

__global__ void k_layernorm(const float* __restrict__ X,
                            const float* __restrict__ prebias,
                            const float* __restrict__ gamma,
                            const float* __restrict__ beta,
                            float* __restrict__ Y, int C, int relu) {
  int node = blockIdx.x * (blockDim.x >> 5) + (threadIdx.x >> 5);
  int lane = threadIdx.x & 31;
  if (node >= NNODES) return;
  const float* x = X + (size_t)node * C;
  float s = 0.0f, sq = 0.0f;
  for (int c = lane; c < C; c += 32) {
    float v = x[c] + (prebias ? prebias[c] : 0.0f);
    s += v; sq += v * v;
  }
#pragma unroll
  for (int m = 16; m >= 1; m >>= 1) {
    s  += __shfl_xor(s,  m, 32);
    sq += __shfl_xor(sq, m, 32);
  }
  float inv  = 1.0f / (float)C;
  float mean = s * inv;
  float var  = sq * inv - mean * mean;
  float rstd = rsqrtf(var + 1e-5f);
  float* y = Y + (size_t)node * C;
  for (int c = lane; c < C; c += 32) {
    float v = x[c] + (prebias ? prebias[c] : 0.0f);
    float o = (v - mean) * rstd * gamma[c] + beta[c];
    if (relu) o = fmaxf(o, 0.0f);
    y[c] = o;
  }
}

// ------------------------------------------------- GAT attention scores
// a_s[n,h] = sum_d h[n,h,d]*att_src[h,d];  a_d likewise.

__global__ void k_att(const float* __restrict__ H,
                      const float* __restrict__ asw,
                      const float* __restrict__ adw,
                      float* __restrict__ as_, float* __restrict__ ad_, int C) {
  int i = blockIdx.x * blockDim.x + threadIdx.x;
  if (i >= NNODES * NHEADS) return;
  int n = i >> 3, h = i & 7;
  int D = C >> 3;
  const float* hp = H + (size_t)n * C + h * D;
  const float* ws = asw + h * D;
  const float* wd = adw + h * D;
  float s = 0.0f, d = 0.0f;
  for (int t = 0; t < D; ++t) { float v = hp[t]; s += v * ws[t]; d += v * wd[t]; }
  as_[i] = s; ad_[i] = d;
}

// ------------------------------------------------- edge pass 1: leaky + segment max

__global__ void k_edge_max(const int* __restrict__ ei,
                           const float* __restrict__ as_,
                           const float* __restrict__ ad_,
                           float* __restrict__ ebuf, float* __restrict__ emax) {
  int i = blockIdx.x * blockDim.x + threadIdx.x;
  if (i >= NETOT * NHEADS) return;
  int e = i >> 3, h = i & 7;
  int s, d; edge_sd(ei, e, s, d);
  float v = as_[s * NHEADS + h] + ad_[d * NHEADS + h];
  v = (v > 0.0f) ? v : 0.2f * v;          // leaky_relu slope 0.2
  ebuf[i] = v;
  atomicMaxF(&emax[d * NHEADS + h], v);
}

// ------------------------------------------------- edge pass 2: exp + segment sum

__global__ void k_edge_exp(const int* __restrict__ ei,
                           float* __restrict__ ebuf,
                           const float* __restrict__ emax,
                           float* __restrict__ den) {
  int i = blockIdx.x * blockDim.x + threadIdx.x;
  if (i >= NETOT * NHEADS) return;
  int e = i >> 3, h = i & 7;
  int s, d; edge_sd(ei, e, s, d);
  (void)s;
  float ee = __expf(ebuf[i] - emax[d * NHEADS + h]);
  ebuf[i] = ee;
  atomicAdd(&den[d * NHEADS + h], ee);
}

// ------------------------------------------------- edge pass 3: weighted scatter
// out[dst] += h[src] * alpha ; accumulator lives in L2 (<=51MB of 192MB).

__global__ void k_edge_agg(const int* __restrict__ ei,
                           const float* __restrict__ ebuf,
                           const float* __restrict__ den,
                           const float* __restrict__ H,
                           float* __restrict__ G, int C) {
  int i = blockIdx.x * blockDim.x + threadIdx.x;
  if (i >= NETOT * NHEADS) return;
  int e = i >> 3, h = i & 7;
  int s, d; edge_sd(ei, e, s, d);
  float alpha = ebuf[i] / den[d * NHEADS + h];
  int D = C >> 3;
  const float* hs = H + (size_t)s * C + h * D;
  float* gd = G + (size_t)d * C + h * D;
  for (int t = 0; t < D; ++t) atomicAdd(&gd[t], hs[t] * alpha);
}

// ------------------------------------------------- tiny head GEMM (Nc=4)

__global__ void k_head2(const float* __restrict__ X,
                        const float* __restrict__ W,
                        const float* __restrict__ b,
                        float* __restrict__ out) {
  int i = blockIdx.x * blockDim.x + threadIdx.x;
  if (i >= NNODES * 4) return;
  int n = i >> 2, o = i & 3;
  float s = b[o];
  const float* x = X + (size_t)n * 32;
#pragma unroll
  for (int k = 0; k < 32; ++k) s += x[k] * W[k * 4 + o];
  out[i] = s;
}

// ================================================================ host side

static inline int cdiv(int a, int b) { return (a + b - 1) / b; }

extern "C" void kernel_launch(void* const* d_in, const int* in_sizes, int n_in,
                              void* d_out, int out_size, void* d_ws, size_t ws_size,
                              hipStream_t stream) {
  (void)in_sizes; (void)n_in; (void)out_size; (void)ws_size;

  const float* x  = (const float*)d_in[0];
  const int*   ei = (const int*)d_in[1];
  auto F = [&](int i) { return (const float*)d_in[i]; };

  // ---- workspace carve (bump allocator, 256B aligned)
  char* wp_ = (char*)d_ws;
  auto carve = [&](size_t bytes) -> void* {
    void* r = (void*)wp_;
    wp_ += (bytes + 255) & ~(size_t)255;
    return r;
  };
  float* bufA  = (float*)carve((size_t)NNODES * 512 * 4);
  float* bufB  = (float*)carve((size_t)NNODES * 512 * 4);
  float* bufH  = (float*)carve((size_t)NNODES * 256 * 4);
  float* bufG  = (float*)carve((size_t)NNODES * 256 * 4);
  float* skip1 = (float*)carve((size_t)NNODES * 64 * 4);
  float* skip2 = (float*)carve((size_t)NNODES * 128 * 4);
  float* skip3 = (float*)carve((size_t)NNODES * 256 * 4);
  unsigned short* Abf = (unsigned short*)carve((size_t)NNODES * 512 * 2);
  unsigned short* Wpk = (unsigned short*)carve((size_t)512 * 512 * 2);
  float* asb  = (float*)carve((size_t)NNODES * NHEADS * 4);
  float* adb  = (float*)carve((size_t)NNODES * NHEADS * 4);
  float* emax = (float*)carve((size_t)NNODES * NHEADS * 4);
  float* den  = (float*)carve((size_t)NNODES * NHEADS * 4);
  float* ebuf = (float*)carve((size_t)NETOT * NHEADS * 4);

  // ---- building blocks
  auto lin = [&](const float* X1, int c1, const float* X2, int c2,
                 const float* W, const float* bias, float* out, int Nc, int relu) {
    int K  = c1 + c2;
    int Kp = (K + 31) & ~31;
    int t1 = NNODES * Kp;
    k_to_bf16<<<cdiv(t1, 256), 256, 0, stream>>>(X1, c1, X2, c2, Abf, Kp);
    int t2 = Kp * Nc;
    k_pack_w<<<cdiv(t2, 256), 256, 0, stream>>>(W, Wpk, K, Kp, Nc);
    int tiles = (NNODES / 16) * (Nc / 16);
    k_gemm_bf16_wmma<<<cdiv(tiles, 8), 256, 0, stream>>>(Abf, Wpk, bias, out, Kp, Nc, tiles, relu);
  };

  auto ln = [&](const float* X, const float* pb, const float* g, const float* b,
                float* Y, int C, int relu) {
    k_layernorm<<<cdiv(NNODES, 8), 256, 0, stream>>>(X, pb, g, b, Y, C, relu);
  };

  auto gat = [&](const float* Xin, int C, int base, float* xout) {
    lin(Xin, C, nullptr, 0, F(base + 4), nullptr, bufH, C, 0);     // h = x @ gat_w
    k_att<<<cdiv(NNODES * NHEADS, 256), 256, 0, stream>>>(bufH, F(base + 5), F(base + 6), asb, adb, C);
    k_fill<<<cdiv(NNODES * NHEADS, 256), 256, 0, stream>>>(emax, -1e30f, NNODES * NHEADS);
    k_fill<<<cdiv(NNODES * NHEADS, 256), 256, 0, stream>>>(den, 0.0f, NNODES * NHEADS);
    k_fill<<<cdiv(NNODES * C, 256), 256, 0, stream>>>(bufG, 0.0f, NNODES * C);
    int et = NETOT * NHEADS;
    k_edge_max<<<cdiv(et, 256), 256, 0, stream>>>(ei, asb, adb, ebuf, emax);
    k_edge_exp<<<cdiv(et, 256), 256, 0, stream>>>(ei, ebuf, emax, den);
    k_edge_agg<<<cdiv(et, 256), 256, 0, stream>>>(ei, ebuf, den, bufH, bufG, C);
    ln(bufG, F(base + 7), F(base + 8), F(base + 9), xout, C, 0);   // ln2(gat + gat_b)
  };

  auto block = [&](const float* xin, int c1, const float* skip, int c2,
                   int base, int cout, float* xout) {
    lin(xin, c1, skip, c2, F(base + 0), F(base + 1), bufB, cout, 0);
    ln(bufB, nullptr, F(base + 2), F(base + 3), bufB, cout, 1);    // relu(ln1(..))
    gat(bufB, cout, base, xout);
  };

  // ---- encoder (param bases: enc0=2, enc1=12, enc2=22)
  block(x,     16,  nullptr, 0, 2,  64,  skip1);
  block(skip1, 64,  nullptr, 0, 12, 128, skip2);
  block(skip2, 128, nullptr, 0, 22, 256, skip3);

  // ---- bottleneck (base 32: w1,b1,g1,be1,w2,b2,g2,be2)
  lin(skip3, 256, nullptr, 0, F(32), F(33), bufB, 512, 0);
  ln(bufB, nullptr, F(34), F(35), bufB, 512, 1);
  lin(bufB, 512, nullptr, 0, F(36), F(37), bufA, 256, 0);
  ln(bufA, nullptr, F(38), F(39), bufA, 256, 1);

  // ---- decoder with skip concat (bases 40, 50, 60)
  block(bufA, 256, skip3, 256, 40, 128, bufA);
  block(bufA, 128, skip2, 128, 50, 64,  bufA);
  block(bufA, 64,  skip1, 64,  60, 64,  bufA);

  // ---- head (base 70: w1,b1,w2,b2) -> [N,4] f32
  lin(bufA, 64, nullptr, 0, F(70), F(71), bufB, 32, 1);            // relu fused
  k_head2<<<cdiv(NNODES * 4, 256), 256, 0, stream>>>(bufB, F(72), F(73), (float*)d_out);
}
